// text_wiki_GCN_53910429499532
// MI455X (gfx1250) — matrix operand order repaired and checked by reading, not compiled, not run on hardware
//
#include <hip/hip_runtime.h>
#include <math.h>

typedef __attribute__((ext_vector_type(16))) __bf16 v16bf;
typedef __attribute__((ext_vector_type(8)))  __bf16 v8bf;
typedef __attribute__((ext_vector_type(8)))  float  v8f;

#define D 128

// ---------- helpers ----------
static __device__ __forceinline__ float gelu1(float x) {
  // jax.nn.gelu default (approximate=True, tanh form)
  float x3 = x * x * x;
  return 0.5f * x * (1.0f + tanhf(0.7978845608028654f * (x + 0.044715f * x3)));
}

static __device__ __forceinline__ void atomicMaxF(float* a, float v) {
  if (v >= 0.0f) atomicMax((int*)a, __float_as_int(v));
  else           atomicMin((unsigned int*)a, __float_as_uint(v));
}

// ---------- weight repack: f32 [128(k)][128(n)] -> lane-contiguous bf16 B-fragments ----------
// layout: ((nt*4 + kt)*32 + lane)*16 bf16; n=nt*16+(lane&15), h=lane/16: K=kt*32+8h+i (i<8), K=kt*32+16+8h+(i-8) (i>=8)
__global__ void pack_w_kernel(const float* __restrict__ W, __bf16* __restrict__ P) {
  int t = blockIdx.x * blockDim.x + threadIdx.x;   // 0..1023
  if (t >= 1024) return;
  int lane = t & 31;
  int kt   = (t >> 5) & 3;
  int nt   = t >> 7;
  int n  = nt * 16 + (lane & 15);
  int kb = kt * 32 + ((lane < 16) ? 0 : 8);
  __bf16* dst = P + ((size_t)((nt * 4 + kt) * 32 + lane)) * 16;
#pragma unroll
  for (int i = 0; i < 16; ++i) dst[i] = (__bf16)W[(size_t)(kb + i + ((i < 8) ? 0 : 8)) * D + n];
}

// ---------- f32 -> bf16 shadow copy (8 elems / thread) ----------
__global__ void cvt_bf16_kernel(const float* __restrict__ src, __bf16* __restrict__ dst, int n8) {
  int t = blockIdx.x * blockDim.x + threadIdx.x;
  if (t >= n8) return;
  const float4* s = (const float4*)src + (size_t)t * 2;
  float4 a = s[0], b = s[1];
  v8bf o;
  o[0] = (__bf16)a.x; o[1] = (__bf16)a.y; o[2] = (__bf16)a.z; o[3] = (__bf16)a.w;
  o[4] = (__bf16)b.x; o[5] = (__bf16)b.y; o[6] = (__bf16)b.z; o[7] = (__bf16)b.w;
  ((v8bf*)dst)[t] = o;
}

// ---------- degree / norm ----------
__global__ void deg_init_kernel(int* __restrict__ a, int* __restrict__ b, int N) {
  int t = blockIdx.x * blockDim.x + threadIdx.x;
  if (t < N) { a[t] = 1; b[t] = 1; }                // self-loop counts as 1
}
__global__ void deg_count_kernel(const int* __restrict__ dst, int* __restrict__ deg, int E) {
  int t = blockIdx.x * blockDim.x + threadIdx.x;
  if (t < E) atomicAdd(&deg[dst[t]], 1);
}
__global__ void dinv_kernel(const int* __restrict__ deg, float* __restrict__ dinv, int N) {
  int t = blockIdx.x * blockDim.x + threadIdx.x;
  if (t < N) dinv[t] = rsqrtf((float)deg[t]);
}

// ---------- H = A(bf16) @ W  via v_wmma_f32_16x16x32_bf16 ----------
// one wave -> 16 rows x 128 cols; 8 waves / block; zero conversion VALU in the loop
__global__ __launch_bounds__(256) void gemm_bf16_kernel(
    const __bf16* __restrict__ A, const __bf16* __restrict__ P,
    float* __restrict__ H, int nTiles) {
  int wave = threadIdx.x >> 5;
  int lane = threadIdx.x & 31;
  int tile = blockIdx.x * 8 + wave;
  if (tile >= nTiles) return;                      // wave-uniform: EXEC stays all-1 for WMMA
  int rowbase = tile * 16;
  int r = lane & 15;
  const __bf16* arow = A + (size_t)(rowbase + r) * D;
  int kHalf = (lane < 16) ? 0 : 8;

  v8f acc[8] = {};
  const v16bf* bp = (const v16bf*)P;
#pragma unroll
  for (int kt = 0; kt < 4; ++kt) {
    int kb = kt * 32 + kHalf;
    v8bf lo = *(const v8bf*)(arow + kb);           // K = kb..kb+7      (16B load)
    v8bf hi = *(const v8bf*)(arow + kb + 16);      // K = kb+16..kb+23  (16B load)
    v16bf a = __builtin_shufflevector(lo, hi, 0,1,2,3,4,5,6,7,8,9,10,11,12,13,14,15);
#pragma unroll
    for (int nt = 0; nt < 8; ++nt) {
      v16bf b = bp[(size_t)nt * 128 + kt * 32 + lane];
      acc[nt] = __builtin_amdgcn_wmma_f32_16x16x32_bf16(
          false, a, false, b, (short)0, acc[nt], false, false);
    }
  }
  int nl = lane & 15;
  int rb = rowbase + ((lane < 16) ? 0 : 8);
#pragma unroll
  for (int nt = 0; nt < 8; ++nt)
#pragma unroll
    for (int r2 = 0; r2 < 8; ++r2)
      H[(size_t)(rb + r2) * D + nt * 16 + nl] = acc[nt][r2];
}

// ---------- fused diff GEMM: ((T - Wk) @ Ww + bw) -> atomicMax pool ----------
__global__ __launch_bounds__(256) void diff_gemm_pool_kernel(
    const float* __restrict__ T, const float* __restrict__ Wk,
    const __bf16* __restrict__ P, const float* __restrict__ bw,
    const int* __restrict__ batch, float* __restrict__ pool, int nTiles) {
  int wave = threadIdx.x >> 5;
  int lane = threadIdx.x & 31;
  int tile = blockIdx.x * 8 + wave;
  if (tile >= nTiles) return;
  int rowbase = tile * 16;
  int r = lane & 15;
  const float* trow = T  + (size_t)(rowbase + r) * D;
  const float* wrow = Wk + (size_t)(rowbase + r) * D;
  int kHalf = (lane < 16) ? 0 : 8;

  v8f acc[8] = {};
  const v16bf* bp = (const v16bf*)P;
#pragma unroll
  for (int kt = 0; kt < 4; ++kt) {
    int kb = kt * 32 + kHalf;
    float4 t0 = *(const float4*)(trow + kb);      float4 w0 = *(const float4*)(wrow + kb);
    float4 t1 = *(const float4*)(trow + kb + 4);  float4 w1 = *(const float4*)(wrow + kb + 4);
    float4 t2 = *(const float4*)(trow + kb + 16); float4 w2 = *(const float4*)(wrow + kb + 16);
    float4 t3 = *(const float4*)(trow + kb + 20); float4 w3 = *(const float4*)(wrow + kb + 20);
    v16bf a;
    a[0]=(__bf16)(t0.x-w0.x); a[1]=(__bf16)(t0.y-w0.y); a[2]=(__bf16)(t0.z-w0.z); a[3]=(__bf16)(t0.w-w0.w);
    a[4]=(__bf16)(t1.x-w1.x); a[5]=(__bf16)(t1.y-w1.y); a[6]=(__bf16)(t1.z-w1.z); a[7]=(__bf16)(t1.w-w1.w);
    a[8]=(__bf16)(t2.x-w2.x); a[9]=(__bf16)(t2.y-w2.y); a[10]=(__bf16)(t2.z-w2.z); a[11]=(__bf16)(t2.w-w2.w);
    a[12]=(__bf16)(t3.x-w3.x); a[13]=(__bf16)(t3.y-w3.y); a[14]=(__bf16)(t3.z-w3.z); a[15]=(__bf16)(t3.w-w3.w);
#pragma unroll
    for (int nt = 0; nt < 8; ++nt) {
      v16bf b = bp[(size_t)nt * 128 + kt * 32 + lane];
      acc[nt] = __builtin_amdgcn_wmma_f32_16x16x32_bf16(
          false, a, false, b, (short)0, acc[nt], false, false);
    }
  }
  int nl = lane & 15;
  int rb = rowbase + ((lane < 16) ? 0 : 8);
#pragma unroll
  for (int r2 = 0; r2 < 8; ++r2) {
    int row = rb + r2;
    int g = batch[row];
#pragma unroll
    for (int nt = 0; nt < 8; ++nt) {
      int col = nt * 16 + nl;
      atomicMaxF(&pool[(size_t)g * D + col], acc[nt][r2] + bw[col]);
    }
  }
}

// ---------- aggregation ----------
// out[n] = b + dinv[n]^2 * H[n]   (self-loop + bias; writes every element)
__global__ void agg_self_bias_kernel(const float* __restrict__ H, const float* __restrict__ dinv,
                                     const float* __restrict__ b, float* __restrict__ out, int N) {
  int t = blockIdx.x * blockDim.x + threadIdx.x;
  int node = t >> 5, lane = t & 31;
  if (node >= N) return;
  float di = dinv[node];
  float s = di * di;
  int c = lane * 4;
  float4 h  = *(const float4*)(H + (size_t)node * D + c);
  float4 bb = *(const float4*)(b + c);
  float4 o; o.x = bb.x + s*h.x; o.y = bb.y + s*h.y; o.z = bb.z + s*h.z; o.w = bb.w + s*h.w;
  *(float4*)(out + (size_t)node * D + c) = o;
}

// one wave per edge: float4 gather + 4 f32 L2 atomics per lane
__global__ void agg_edge_kernel(const float* __restrict__ H, const float* __restrict__ dinv,
                                const int* __restrict__ src, const int* __restrict__ dst,
                                float* __restrict__ out, int E) {
  int t = blockIdx.x * blockDim.x + threadIdx.x;
  int e = t >> 5, lane = t & 31;
  if (e >= E) return;
  int s = src[e], d = dst[e];
  float nm = dinv[s] * dinv[d];
  int c = lane * 4;
  float4 h = *(const float4*)(H + (size_t)s * D + c);
  float* o = out + (size_t)d * D + c;
  atomicAdd(o + 0, nm * h.x);
  atomicAdd(o + 1, nm * h.y);
  atomicAdd(o + 2, nm * h.z);
  atomicAdd(o + 3, nm * h.w);
}

// in-place f32 GELU; optionally also writes a bf16 shadow copy for the next GEMM
__global__ void gelu_dual_kernel(float* __restrict__ x, __bf16* __restrict__ xbf, int n8) {
  int t = blockIdx.x * blockDim.x + threadIdx.x;
  if (t >= n8) return;
  float4* p = (float4*)x + (size_t)t * 2;
  float4 a = p[0], b = p[1];
  a.x = gelu1(a.x); a.y = gelu1(a.y); a.z = gelu1(a.z); a.w = gelu1(a.w);
  b.x = gelu1(b.x); b.y = gelu1(b.y); b.z = gelu1(b.z); b.w = gelu1(b.w);
  p[0] = a; p[1] = b;
  if (xbf) {
    v8bf o;
    o[0] = (__bf16)a.x; o[1] = (__bf16)a.y; o[2] = (__bf16)a.z; o[3] = (__bf16)a.w;
    o[4] = (__bf16)b.x; o[5] = (__bf16)b.y; o[6] = (__bf16)b.z; o[7] = (__bf16)b.w;
    ((v8bf*)xbf)[t] = o;
  }
}

__global__ void pool_init_kernel(float* __restrict__ pool, int n) {
  int t = blockIdx.x * blockDim.x + threadIdx.x;
  if (t < n) pool[t] = -INFINITY;
}

// ---------- driver ----------
extern "C" void kernel_launch(void* const* d_in, const int* in_sizes, int n_in,
                              void* d_out, int out_size, void* d_ws, size_t ws_size,
                              hipStream_t stream) {
  (void)n_in; (void)out_size; (void)ws_size;
  const float* x    = (const float*)d_in[0];
  const int*   eiT  = (const int*)d_in[1];
  const int*   eiW  = (const int*)d_in[2];
  const int*   batch= (const int*)d_in[3];
  const float* W1t  = (const float*)d_in[4];  const float* b1t = (const float*)d_in[5];
  const float* W2t  = (const float*)d_in[6];  const float* b2t = (const float*)d_in[7];
  const float* W1w  = (const float*)d_in[8];  const float* b1w = (const float*)d_in[9];
  const float* W2w  = (const float*)d_in[10]; const float* b2w = (const float*)d_in[11];
  const float* Ww   = (const float*)d_in[12]; const float* bw  = (const float*)d_in[13];

  const int N = in_sizes[0] / D;          // 50000
  const int E = in_sizes[1] / 2;          // 800000
  const int nTiles = (N + 15) / 16;       // 3125 (N is a multiple of 16)

  const int* srcT = eiT;      const int* dstT = eiT + E;
  const int* srcW = eiW;      const int* dstW = eiW + E;

  // workspace carve-up
  char* ws = (char*)d_ws;
  float* H    = (float*)ws;            ws += (size_t)N * D * sizeof(float);
  float* Tbuf = (float*)ws;            ws += (size_t)N * D * sizeof(float);
  __bf16* Xbf = (__bf16*)ws;           ws += (size_t)N * D * sizeof(__bf16);   // bf16 shadow of x
  __bf16* Abf = (__bf16*)ws;           ws += (size_t)N * D * sizeof(__bf16);   // bf16 shadow of activations
  int*   degT = (int*)ws;              ws += (size_t)N * sizeof(int);
  int*   degW = (int*)ws;              ws += (size_t)N * sizeof(int);
  float* dinvT= (float*)ws;            ws += (size_t)N * sizeof(float);
  float* dinvW= (float*)ws;            ws += (size_t)N * sizeof(float);
  __bf16* pk  = (__bf16*)ws;           // 5 * 16384 bf16 packed weights
  __bf16* pW1t = pk;            __bf16* pW2t = pk + 16384;
  __bf16* pW1w = pk + 2*16384;  __bf16* pW2w = pk + 3*16384;
  __bf16* pWw  = pk + 4*16384;

  float* text = (float*)d_out;
  float* pool = text + (size_t)N * D;

  dim3 b256(256);
  int gN    = (N + 255) / 256;
  int gE    = (E + 255) / 256;
  int gNode = (N * 32 + 255) / 256;     // node*lane passes
  int gEdge = (E * 32 + 255) / 256;     // edge*lane passes
  int n8    = N * D / 8;
  int gElem8= (n8 + 255) / 256;         // 8-elem elementwise passes
  int gGemm = (nTiles + 7) / 8;

  // weight repack + input shadow (one-time per call, tiny)
  pack_w_kernel<<<4, b256, 0, stream>>>(W1t, pW1t);
  pack_w_kernel<<<4, b256, 0, stream>>>(W2t, pW2t);
  pack_w_kernel<<<4, b256, 0, stream>>>(W1w, pW1w);
  pack_w_kernel<<<4, b256, 0, stream>>>(W2w, pW2w);
  pack_w_kernel<<<4, b256, 0, stream>>>(Ww,  pWw);
  cvt_bf16_kernel<<<gElem8, b256, 0, stream>>>(x, Xbf, n8);

  // degrees + symmetric norms
  deg_init_kernel<<<gN, b256, 0, stream>>>(degT, degW, N);
  deg_count_kernel<<<gE, b256, 0, stream>>>(dstT, degT, E);
  deg_count_kernel<<<gE, b256, 0, stream>>>(dstW, degW, E);
  dinv_kernel<<<gN, b256, 0, stream>>>(degT, dinvT, N);
  dinv_kernel<<<gN, b256, 0, stream>>>(degW, dinvW, N);

  // text layer 1: H = x@W1t ; Tbuf = gelu(agg(H)+b1t) ; Abf = bf16(Tbuf)
  gemm_bf16_kernel<<<gGemm, b256, 0, stream>>>(Xbf, pW1t, H, nTiles);
  agg_self_bias_kernel<<<gNode, b256, 0, stream>>>(H, dinvT, b1t, Tbuf, N);
  agg_edge_kernel<<<gEdge, b256, 0, stream>>>(H, dinvT, srcT, dstT, Tbuf, E);
  gelu_dual_kernel<<<gElem8, b256, 0, stream>>>(Tbuf, Abf, n8);

  // text layer 2 -> d_out text region (f32 only; diff GEMM reads f32)
  gemm_bf16_kernel<<<gGemm, b256, 0, stream>>>(Abf, pW2t, H, nTiles);
  agg_self_bias_kernel<<<gNode, b256, 0, stream>>>(H, dinvT, b2t, text, N);
  agg_edge_kernel<<<gEdge, b256, 0, stream>>>(H, dinvT, srcT, dstT, text, E);
  gelu_dual_kernel<<<gElem8, b256, 0, stream>>>(text, nullptr, n8);

  // wiki layer 1 (reuse Tbuf / Abf)
  gemm_bf16_kernel<<<gGemm, b256, 0, stream>>>(Xbf, pW1w, H, nTiles);
  agg_self_bias_kernel<<<gNode, b256, 0, stream>>>(H, dinvW, b1w, Tbuf, N);
  agg_edge_kernel<<<gEdge, b256, 0, stream>>>(H, dinvW, srcW, dstW, Tbuf, E);
  gelu_dual_kernel<<<gElem8, b256, 0, stream>>>(Tbuf, Abf, n8);

  // wiki layer 2 (GEMM consumes Abf fully before agg rewrites Tbuf — stream-ordered)
  gemm_bf16_kernel<<<gGemm, b256, 0, stream>>>(Abf, pW2w, H, nTiles);
  agg_self_bias_kernel<<<gNode, b256, 0, stream>>>(H, dinvW, b2w, Tbuf, N);
  agg_edge_kernel<<<gEdge, b256, 0, stream>>>(H, dinvW, srcW, dstW, Tbuf, E);
  gelu_dual_kernel<<<gElem8, b256, 0, stream>>>(Tbuf, nullptr, n8);

  // fused diff projection + segment-max pool
  pool_init_kernel<<<(128 * D + 255) / 256, b256, 0, stream>>>(pool, 128 * D);
  diff_gemm_pool_kernel<<<gGemm, b256, 0, stream>>>(text, Tbuf, pWw, bw, batch, pool, nTiles);
}
